// ExtendedDecoderLayer_5274219840235
// MI455X (gfx1250) — matrix operand order repaired and checked by reading, not compiled, not run on hardware
//
#include <hip/hip_runtime.h>

typedef __attribute__((ext_vector_type(16))) __bf16 v16bf;
typedef __attribute__((ext_vector_type(8)))  __bf16 v8bf;
typedef __attribute__((ext_vector_type(4)))  __bf16 v4bf;
typedef __attribute__((ext_vector_type(8)))  float  v8f;
typedef __attribute__((ext_vector_type(4)))  unsigned int u32x4;
typedef __attribute__((ext_vector_type(8)))  int i32x8;
typedef __attribute__((ext_vector_type(4)))  int i32x4;

union FragBF {
  v16bf v;
  struct { v8bf lo, hi; } h;
};

__device__ __forceinline__ v8f wmma_bf16(const FragBF& a, const FragBF& b, v8f c) {
  return __builtin_amdgcn_wmma_f32_16x16x32_bf16(false, a.v, false, b.v,
                                                 (short)0, c, false, false);
}

__device__ __forceinline__ v8f v8f_zero() {
  v8f z = {0.f, 0.f, 0.f, 0.f, 0.f, 0.f, 0.f, 0.f};
  return z;
}

// ---------------------------------------------------------------------------
// Tensor Data Mover: DMA a GBM x GBK bf16 tile (tile start (row0,col0) inside
// a [nrows x ncols] row-major bf16 tensor) into LDS at byte address `lds`,
// inserting 4 DWORDs of padding every 16 DWORDs (row: 32 bf16 = 16 DW) so the
// landed LDS row stride is 40 bf16 = 80 B (16 B aligned) for fragment loads.
// D# layout per CDNA5 ISA 8.3/8.4; inline asm form is portable across
// toolchains (builtin arity differs between ROCm 7.2 and amdgpu-toolchain).
// ---------------------------------------------------------------------------
#define GBM 128
#define GBN 128
#define GBK 32
#define GLD 40   // bf16 LDS row stride (32 + 4DW pad)

__device__ __forceinline__ void tdm_load_panel(const __bf16* g, int row0, int col0,
                                               int nrows, int ncols, unsigned lds)
{
  unsigned long long ga = (unsigned long long)g +
      ((unsigned long long)row0 * (unsigned)ncols + (unsigned)col0) * 2ull;
  u32x4 g0;
  g0[0] = 1u;                                               // count=1 (valid D#)
  g0[1] = lds;                                              // lds_addr
  g0[2] = (unsigned)ga;                                     // global_addr lo
  g0[3] = (unsigned)((ga >> 32) & 0x01FFFFFFull) | (2u << 30);  // addr hi | type=2
  i32x8 g1;
  g1[0] = (1 << 16)        // data_size = 1 (2 bytes)
        | (1 << 20)        // pad_enable
        | (3 << 22)        // pad_interval: 16 DWORDs
        | (3 << 25);       // pad_amount: 4 DWORDs
  g1[1] = (ncols & 0xFFFF) << 16;                           // tensor_dim0 lo16
  g1[2] = ((ncols >> 16) & 0xFFFF) | ((nrows & 0xFFFF) << 16);  // dim0 hi | dim1 lo
  g1[3] = ((nrows >> 16) & 0xFFFF) | (GBK << 16);           // dim1 hi | tile_dim0
  g1[4] = GBM;                                              // tile_dim1 (tile_dim2=0)
  g1[5] = ncols;                                            // tensor_dim0_stride lo
  g1[6] = 0; g1[7] = 0;
  i32x4 gz; gz[0] = gz[1] = gz[2] = gz[3] = 0;              // 2D: groups 2/3 zero
  asm volatile("tensor_load_to_lds %0, %1, %2, %3"
               :: "s"(g0), "s"(g1), "s"(gz), "s"(gz)
               : "memory");
}

// ---------------------------------------------------------------------------
// Elementwise f32 -> bf16 converter (float4 -> 4x bf16 per thread).
// ---------------------------------------------------------------------------
__global__ __launch_bounds__(256) void cvt_bf16_kernel(
    const float* __restrict__ X, __bf16* __restrict__ Y, int n4)
{
  const int i = blockIdx.x * 256 + threadIdx.x;
  if (i >= n4) return;
  const float4 f = ((const float4*)X)[i];
  v4bf o;
  o[0] = (__bf16)f.x; o[1] = (__bf16)f.y; o[2] = (__bf16)f.z; o[3] = (__bf16)f.w;
  ((v4bf*)Y)[i] = o;
}

// ---------------------------------------------------------------------------
// Tiled GEMM: C[M,N] = act(A[M,K] @ W[N,K]^T + bias[N]); A/W bf16 in memory,
// f32 accumulate. Block = 256 thr (8 waves), tile 128x128x32; TDM double-
// buffered LDS staging (wave 0 issues, TENSORcnt + barrier sync); each wave
// computes a 32x64 subtile = 8 v_wmma_f32_16x16x32_bf16 per k-step.
// ---------------------------------------------------------------------------
__global__ __launch_bounds__(256) void gemm_bias_kernel(
    const __bf16* __restrict__ A, const __bf16* __restrict__ W,
    const float* __restrict__ bias, float* __restrict__ C,
    int M, int N, int K, int relu)
{
  __shared__ __bf16 As[2][GBM * GLD];
  __shared__ __bf16 Bs[2][GBN * GLD];

  const int tid   = threadIdx.x;
  const int wave  = tid >> 5;
  const int lane  = tid & 31;
  const int lhalf = lane >> 4;
  const int l16   = lane & 15;
  const int m0 = blockIdx.y * GBM;
  const int n0 = blockIdx.x * GBN;
  const int wm = (wave >> 1) * 32;   // wave grid 4 (M) x 2 (N)
  const int wn = (wave & 1) * 64;

  const unsigned ldsA[2] = { (unsigned)(unsigned long long)&As[0][0],
                             (unsigned)(unsigned long long)&As[1][0] };
  const unsigned ldsB[2] = { (unsigned)(unsigned long long)&Bs[0][0],
                             (unsigned)(unsigned long long)&Bs[1][0] };

  v8f acc[2][4];
#pragma unroll
  for (int i = 0; i < 2; ++i)
#pragma unroll
    for (int j = 0; j < 4; ++j) acc[i][j] = v8f_zero();

  // prologue: DMA first panels into buffer 0
  if (tid < 32) {
    tdm_load_panel(A, m0, 0, M, K, ldsA[0]);
    tdm_load_panel(W, n0, 0, N, K, ldsB[0]);
    __builtin_amdgcn_s_wait_tensorcnt(0);
  }
  __syncthreads();

  int buf = 0;
  for (int kc = 0; kc < K; kc += GBK) {
    if (tid < 32 && kc + GBK < K) {   // overlap next-panel DMA with compute
      tdm_load_panel(A, m0, kc + GBK, M, K, ldsA[buf ^ 1]);
      tdm_load_panel(W, n0, kc + GBK, N, K, ldsB[buf ^ 1]);
    }

    FragBF af[2], bfr[4];
#pragma unroll
    for (int i = 0; i < 2; ++i) {     // A frag: row = l16, K chunks hi8 / 16+hi8
      const __bf16* p = &As[buf][(wm + i * 16 + l16) * GLD + lhalf * 8];
      af[i].h.lo = *(const v8bf*)p;
      af[i].h.hi = *(const v8bf*)(p + 16);
    }
#pragma unroll
    for (int j = 0; j < 4; ++j) {     // B frag: col = l16, K chunks hi8 / 16+hi8
      const __bf16* p = &Bs[buf][(wn + j * 16 + l16) * GLD + lhalf * 8];
      bfr[j].h.lo = *(const v8bf*)p;
      bfr[j].h.hi = *(const v8bf*)(p + 16);
    }
#pragma unroll
    for (int i = 0; i < 2; ++i)
#pragma unroll
      for (int j = 0; j < 4; ++j)
        acc[i][j] = wmma_bf16(af[i], bfr[j], acc[i][j]);

    if (tid < 32) __builtin_amdgcn_s_wait_tensorcnt(0);
    __syncthreads();
    buf ^= 1;
  }

#pragma unroll
  for (int i = 0; i < 2; ++i) {
    const int row = m0 + wm + i * 16 + lhalf * 8;   // VGPR v -> M = v (+8 hi half)
#pragma unroll
    for (int j = 0; j < 4; ++j) {
      const int col = n0 + wn + j * 16 + l16;
      const float bv = bias[col];
#pragma unroll
      for (int v = 0; v < 8; ++v) {
        float r = acc[i][j][v] + bv;
        if (relu) r = fmaxf(r, 0.f);
        C[(size_t)(row + v) * N + col] = r;
      }
    }
  }
}

// ---------------------------------------------------------------------------
// P[b,h,q,r] = dot(Q[b,q,h*64:...], rel_k[r]) — one thread per (b,h,q,r).
// ---------------------------------------------------------------------------
__global__ __launch_bounds__(256) void relproj_kernel(
    const float* __restrict__ Q, const float* __restrict__ relk,
    float* __restrict__ P)
{
  const size_t idx = (size_t)blockIdx.x * 256 + threadIdx.x;  // B*H*L*32
  const int r = (int)(idx & 31);
  size_t t = idx >> 5;
  const int q = (int)(t & 1023); t >>= 10;
  const int h = (int)(t & 15);
  const int b = (int)(t >> 4);
  const float* qp = Q + ((size_t)b * 1024 + q) * 1024 + h * 64;
  const float* rp = relk + r * 64;
  float s = 0.f;
#pragma unroll
  for (int d = 0; d < 64; ++d) s += qp[d] * rp[d];
  P[idx] = s;
}

// ---------------------------------------------------------------------------
// Flash-style relative attention. Block = 128 thr (4 waves), 64 q-rows per
// block, 32-key chunks. scores = (Q·K^T + P[q,rid] + rel_b[rid,h]) / 8,
// causal mask optional, online softmax, O = softmax(S)·V.
// ---------------------------------------------------------------------------
__global__ __launch_bounds__(128) void attn_rel_kernel(
    const float* __restrict__ Qm, const float* __restrict__ Km,
    const float* __restrict__ Vm, const float* __restrict__ Pm,
    const int* __restrict__ rel_ids, const float* __restrict__ rel_b,
    float* __restrict__ Out, int causal)
{
  constexpr int L = 1024, Dm = 1024, Hh = 16;
  constexpr int SQ = 72, SV = 40;
  __shared__ __bf16 Qs[64 * SQ];
  __shared__ __bf16 Ks[32 * SQ];
  __shared__ __bf16 Vt[64 * SV];   // V transposed: [d][key]
  __shared__ __bf16 Ps[64 * SV];   // probs staging: [local q row][key]
  __shared__ float  Pb[64 * 32];
  __shared__ float  rbs[32];

  const int tid   = threadIdx.x;
  const int wave  = tid >> 5, lane = tid & 31;
  const int lhalf = lane >> 4, l16 = lane & 15;
  const int q0 = blockIdx.x * 64;
  const int h  = blockIdx.y;
  const int b  = blockIdx.z;

#pragma unroll
  for (int i = 0; i < 32; ++i) {   // stage Q tile 64x64 -> bf16
    const int idx = tid + i * 128;
    const int r = idx >> 6, c = idx & 63;
    Qs[r * SQ + c] = (__bf16)Qm[((size_t)b * L + q0 + r) * Dm + h * 64 + c];
  }
#pragma unroll
  for (int i = 0; i < 16; ++i) {   // stage P 64x32
    const int idx = tid + i * 128;
    const int r = idx >> 5;
    Pb[idx] = Pm[(((size_t)b * Hh + h) * L + q0 + r) * 32 + (idx & 31)];
  }
  if (tid < 32) rbs[tid] = rel_b[tid * Hh + h];
  __syncthreads();

  FragBF aq[2];
#pragma unroll
  for (int kk = 0; kk < 2; ++kk) {
    const __bf16* p = &Qs[(wave * 16 + l16) * SQ + kk * 32 + lhalf * 8];
    aq[kk].h.lo = *(const v8bf*)p;
    aq[kk].h.hi = *(const v8bf*)(p + 16);
  }

  float mrow[8], lrow[8];
  v8f o[4];
#pragma unroll
  for (int v = 0; v < 8; ++v) { mrow[v] = -1e30f; lrow[v] = 0.f; }
#pragma unroll
  for (int j = 0; j < 4; ++j) o[j] = v8f_zero();

  const int kend = causal ? (q0 + 64) : L;
  for (int k0 = 0; k0 < kend; k0 += 32) {
    __syncthreads();
    if (k0 + 32 < kend) {  // prefetch next K/V chunk rows
      const int pr = tid >> 2, pc = (tid & 3) * 16;
      const size_t pbase = ((size_t)b * L + k0 + 32 + pr) * Dm + h * 64 + pc;
      __builtin_prefetch(Km + pbase, 0, 1);
      __builtin_prefetch(Vm + pbase, 0, 1);
    }
#pragma unroll
    for (int i = 0; i < 16; ++i) {  // stage K (row-major) and V (transposed)
      const int idx = tid + i * 128;
      const int r = idx >> 6, c = idx & 63;
      const size_t base = ((size_t)b * L + k0 + r) * Dm + h * 64 + c;
      Ks[r * SQ + c] = (__bf16)Km[base];
      Vt[c * SV + r] = (__bf16)Vm[base];
    }
    __syncthreads();

    v8f s[2];
#pragma unroll
    for (int t = 0; t < 2; ++t) {   // two 16-key score tiles, DH=64 => 2 k-steps
      s[t] = v8f_zero();
#pragma unroll
      for (int kk = 0; kk < 2; ++kk) {
        FragBF bk;
        const __bf16* p = &Ks[(t * 16 + l16) * SQ + kk * 32 + lhalf * 8];
        bk.h.lo = *(const v8bf*)p;
        bk.h.hi = *(const v8bf*)(p + 16);
        s[t] = wmma_bf16(aq[kk], bk, s[t]);
      }
    }

#pragma unroll
    for (int t = 0; t < 2; ++t) {   // relative terms, scale, mask
      const int key = k0 + t * 16 + l16;
#pragma unroll
      for (int v = 0; v < 8; ++v) {
        const int lr = wave * 16 + lhalf * 8 + v;
        const int qrow = q0 + lr;
        const int rid = rel_ids[(size_t)qrow * L + key];
        float sv = s[t][v] + Pb[lr * 32 + rid] + rbs[rid];
        sv *= 0.125f;
        if (causal && key > qrow) sv = -1e9f;
        s[t][v] = sv;
      }
    }

#pragma unroll
    for (int v = 0; v < 8; ++v) {   // online softmax update
      float mx = fmaxf(s[0][v], s[1][v]);
#pragma unroll
      for (int off = 8; off > 0; off >>= 1)
        mx = fmaxf(mx, __shfl_xor(mx, off, 32));
      const float mnew = fmaxf(mrow[v], mx);
      const float corr = __expf(mrow[v] - mnew);
      const float p0 = __expf(s[0][v] - mnew);
      const float p1 = __expf(s[1][v] - mnew);
      float rs = p0 + p1;
#pragma unroll
      for (int off = 8; off > 0; off >>= 1)
        rs += __shfl_xor(rs, off, 32);
      lrow[v] = lrow[v] * corr + rs;
      mrow[v] = mnew;
#pragma unroll
      for (int j = 0; j < 4; ++j) o[j][v] *= corr;
      const int lr = wave * 16 + lhalf * 8 + v;
      Ps[lr * SV + 0 * 16 + l16] = (__bf16)p0;
      Ps[lr * SV + 1 * 16 + l16] = (__bf16)p1;
    }

    FragBF ap;  // re-layout probs C-tile -> A-fragment via LDS (wave-private)
    const __bf16* pp = &Ps[(wave * 16 + l16) * SV + lhalf * 8];
    ap.h.lo = *(const v8bf*)pp;
    ap.h.hi = *(const v8bf*)(pp + 16);
#pragma unroll
    for (int j = 0; j < 4; ++j) {   // O += P(16x32) x V(32x16) per d-tile
      FragBF bv;
      const __bf16* p = &Vt[(j * 16 + l16) * SV + lhalf * 8];
      bv.h.lo = *(const v8bf*)p;
      bv.h.hi = *(const v8bf*)(p + 16);
      o[j] = wmma_bf16(ap, bv, o[j]);
    }
  }

#pragma unroll
  for (int v = 0; v < 8; ++v) {
    const float inv = 1.f / lrow[v];
    const int qrow = q0 + wave * 16 + lhalf * 8 + v;
#pragma unroll
    for (int j = 0; j < 4; ++j)
      Out[((size_t)b * L + qrow) * Dm + h * 64 + j * 16 + l16] = o[j][v] * inv;
  }
}

// ---------------------------------------------------------------------------
// Y = LayerNorm(X + A) * g + be ; one block per row (D = 1024).
// ---------------------------------------------------------------------------
__global__ __launch_bounds__(256) void add_ln_kernel(
    const float* __restrict__ X, const float* __restrict__ A,
    const float* __restrict__ g, const float* __restrict__ be,
    float* __restrict__ Y)
{
  const int row = blockIdx.x;
  const int tid = threadIdx.x;
  const float* x = X + (size_t)row * 1024;
  const float* a = A + (size_t)row * 1024;
  float v[4], s = 0.f, ss = 0.f;
#pragma unroll
  for (int i = 0; i < 4; ++i) {
    const int c = tid + i * 256;
    v[i] = x[c] + a[c];
    s += v[i]; ss += v[i] * v[i];
  }
#pragma unroll
  for (int off = 16; off > 0; off >>= 1) {
    s  += __shfl_xor(s,  off, 32);
    ss += __shfl_xor(ss, off, 32);
  }
  __shared__ float sums[8], sqs[8];
  const int wave = tid >> 5, lane = tid & 31;
  if (lane == 0) { sums[wave] = s; sqs[wave] = ss; }
  __syncthreads();
  float ts = 0.f, tss = 0.f;
#pragma unroll
  for (int i = 0; i < 8; ++i) { ts += sums[i]; tss += sqs[i]; }
  const float mean = ts * (1.f / 1024.f);
  const float var  = tss * (1.f / 1024.f) - mean * mean;
  const float rstd = rsqrtf(var + 1e-5f);
#pragma unroll
  for (int i = 0; i < 4; ++i) {
    const int c = tid + i * 256;
    Y[(size_t)row * 1024 + c] = (v[i] - mean) * rstd * g[c] + be[c];
  }
}

// ---------------------------------------------------------------------------
extern "C" void kernel_launch(void* const* d_in, const int* in_sizes, int n_in,
                              void* d_out, int out_size, void* d_ws, size_t ws_size,
                              hipStream_t stream)
{
  (void)in_sizes; (void)n_in; (void)out_size; (void)ws_size;
  const float* x    = (const float*)d_in[0];
  const float* enc  = (const float*)d_in[1];
  const int*   rel1 = (const int*)d_in[2];
  const int*   rel2 = (const int*)d_in[3];
  // d_in[4]/d_in[5]: causal-tril / all-true masks, encoded via `causal` flag.
  const float *wq1=(const float*)d_in[6],  *wk1=(const float*)d_in[7];
  const float *wv1=(const float*)d_in[8],  *wo1=(const float*)d_in[9];
  const float *bq1=(const float*)d_in[10], *bk1=(const float*)d_in[11];
  const float *bv1=(const float*)d_in[12], *bo1=(const float*)d_in[13];
  const float *relk1=(const float*)d_in[14], *relb1=(const float*)d_in[15];
  const float *wq2=(const float*)d_in[16], *wk2=(const float*)d_in[17];
  const float *wv2=(const float*)d_in[18], *wo2=(const float*)d_in[19];
  const float *bq2=(const float*)d_in[20], *bk2=(const float*)d_in[21];
  const float *bv2=(const float*)d_in[22], *bo2=(const float*)d_in[23];
  const float *relk2=(const float*)d_in[24], *relb2=(const float*)d_in[25];
  const float *w1=(const float*)d_in[26], *b1=(const float*)d_in[27];
  const float *w2=(const float*)d_in[28], *b2=(const float*)d_in[29];
  const float *g1=(const float*)d_in[30], *be1=(const float*)d_in[31];
  const float *g2=(const float*)d_in[32], *be2=(const float*)d_in[33];
  const float *g3=(const float*)d_in[34], *be3=(const float*)d_in[35];

  // ---- workspace layout (byte offsets; aggressive reuse) ----
  char* base = (char*)d_ws;
  const size_t MB = 1024 * 1024;
  float*  Qb  = (float*)(base +   0 * MB);   // 16MB, also O-proj output
  float*  Kb  = (float*)(base +  16 * MB);   // 16MB, also FF2 output
  float*  Vb  = (float*)(base +  32 * MB);   // 16MB
  float*  Ab  = (float*)(base +  48 * MB);   // 16MB attention out (f32)
  float*  X1  = (float*)(base +  64 * MB);   // 16MB
  float*  X2  = (float*)(base +  80 * MB);   // 16MB
  float*  Pf  = (float*)(base +  96 * MB);   // 8MB rel projections
  __bf16* Abb = (__bf16*)(base + 96 * MB);   // 8MB, reuses Pf AFTER attention
  float*  Fb  = (float*)(base + 112 * MB);   // 64MB FFN hidden (f32)
  __bf16* Fbb = (__bf16*)(base + 32 * MB);   // 32MB, reuses Vb+Ab during FFN
  __bf16* wqb1=(__bf16*)(base+176*MB), *wkb1=(__bf16*)(base+178*MB);
  __bf16* wvb1=(__bf16*)(base+180*MB), *wob1=(__bf16*)(base+182*MB);
  __bf16* wqb2=(__bf16*)(base+184*MB), *wkb2=(__bf16*)(base+186*MB);
  __bf16* wvb2=(__bf16*)(base+188*MB), *wob2=(__bf16*)(base+190*MB);
  __bf16* w1b =(__bf16*)(base+192*MB);       // 8MB
  __bf16* w2b =(__bf16*)(base+200*MB);       // 8MB
  __bf16* xb  =(__bf16*)(base+208*MB);       // 8MB; X1b reuses after self block
  __bf16* encb=(__bf16*)(base+216*MB);       // 8MB; X2b reuses after cross block
  __bf16* X1b = xb;
  __bf16* X2b = encb;
  float*  Ob  = Qb;

  const dim3 gp(1024 / 128, 4096 / 128);   // N=1024 GEMMs
  const dim3 gf(4096 / 128, 4096 / 128);   // N=4096 GEMM (FFN up)
  const dim3 ga(16, 16, 4);
  const int cvt1M = (1024 * 1024 / 4) / 256;   // D*D elements
  const int cvt4M = 4 * cvt1M;                 // B*L*D or FF*D
  const int cvt16M = 16 * cvt1M;               // B*L*FF

  // ---- one-time (per launch) bf16 mirrors ----
  cvt_bf16_kernel<<<cvt1M, 256, 0, stream>>>(wq1, wqb1, 1024*1024/4);
  cvt_bf16_kernel<<<cvt1M, 256, 0, stream>>>(wk1, wkb1, 1024*1024/4);
  cvt_bf16_kernel<<<cvt1M, 256, 0, stream>>>(wv1, wvb1, 1024*1024/4);
  cvt_bf16_kernel<<<cvt1M, 256, 0, stream>>>(wo1, wob1, 1024*1024/4);
  cvt_bf16_kernel<<<cvt1M, 256, 0, stream>>>(wq2, wqb2, 1024*1024/4);
  cvt_bf16_kernel<<<cvt1M, 256, 0, stream>>>(wk2, wkb2, 1024*1024/4);
  cvt_bf16_kernel<<<cvt1M, 256, 0, stream>>>(wv2, wvb2, 1024*1024/4);
  cvt_bf16_kernel<<<cvt1M, 256, 0, stream>>>(wo2, wob2, 1024*1024/4);
  cvt_bf16_kernel<<<cvt4M, 256, 0, stream>>>(w1, w1b, 4096*1024/4);
  cvt_bf16_kernel<<<cvt4M, 256, 0, stream>>>(w2, w2b, 4096*1024/4);
  cvt_bf16_kernel<<<cvt4M, 256, 0, stream>>>(x,   xb,   4096*1024/4);
  cvt_bf16_kernel<<<cvt4M, 256, 0, stream>>>(enc, encb, 4096*1024/4);

  // ---- self attention ----
  gemm_bias_kernel<<<gp, 256, 0, stream>>>(xb, wqb1, bq1, Qb, 4096, 1024, 1024, 0);
  gemm_bias_kernel<<<gp, 256, 0, stream>>>(xb, wkb1, bk1, Kb, 4096, 1024, 1024, 0);
  gemm_bias_kernel<<<gp, 256, 0, stream>>>(xb, wvb1, bv1, Vb, 4096, 1024, 1024, 0);
  relproj_kernel<<<8192, 256, 0, stream>>>(Qb, relk1, Pf);
  attn_rel_kernel<<<ga, 128, 0, stream>>>(Qb, Kb, Vb, Pf, rel1, relb1, Ab, 1);
  cvt_bf16_kernel<<<cvt4M, 256, 0, stream>>>(Ab, Abb, 4096*1024/4);
  gemm_bias_kernel<<<gp, 256, 0, stream>>>(Abb, wob1, bo1, Ob, 4096, 1024, 1024, 0);
  add_ln_kernel<<<4096, 256, 0, stream>>>(x, Ob, g1, be1, X1);
  cvt_bf16_kernel<<<cvt4M, 256, 0, stream>>>(X1, X1b, 4096*1024/4);

  // ---- cross attention ----
  gemm_bias_kernel<<<gp, 256, 0, stream>>>(X1b,  wqb2, bq2, Qb, 4096, 1024, 1024, 0);
  gemm_bias_kernel<<<gp, 256, 0, stream>>>(encb, wkb2, bk2, Kb, 4096, 1024, 1024, 0);
  gemm_bias_kernel<<<gp, 256, 0, stream>>>(encb, wvb2, bv2, Vb, 4096, 1024, 1024, 0);
  relproj_kernel<<<8192, 256, 0, stream>>>(Qb, relk2, Pf);
  attn_rel_kernel<<<ga, 128, 0, stream>>>(Qb, Kb, Vb, Pf, rel2, relb2, Ab, 0);
  cvt_bf16_kernel<<<cvt4M, 256, 0, stream>>>(Ab, Abb, 4096*1024/4);
  gemm_bias_kernel<<<gp, 256, 0, stream>>>(Abb, wob2, bo2, Ob, 4096, 1024, 1024, 0);
  add_ln_kernel<<<4096, 256, 0, stream>>>(X1, Ob, g2, be2, X2);
  cvt_bf16_kernel<<<cvt4M, 256, 0, stream>>>(X2, X2b, 4096*1024/4);

  // ---- feed-forward ----
  gemm_bias_kernel<<<gf, 256, 0, stream>>>(X2b, w1b, b1, Fb, 4096, 4096, 1024, 1);
  cvt_bf16_kernel<<<cvt16M, 256, 0, stream>>>(Fb, Fbb, 4096*4096/4);
  gemm_bias_kernel<<<gp, 256, 0, stream>>>(Fbb, w2b, b2, Kb, 4096, 1024, 4096, 0);
  add_ln_kernel<<<4096, 256, 0, stream>>>(X2, Kb, g3, be3, (float*)d_out);
}
